// EdgePredictor_61100204753674
// MI455X (gfx1250) — matrix operand order repaired and checked by reading, not compiled, not run on hardware
//
#include <hip/hip_runtime.h>
#include <hip/hip_bf16.h>
#include <stdint.h>

#define N_NODES 100000
#define N_EDGES 1600000
#define IN_CH   128
#define HID     64

typedef __attribute__((ext_vector_type(16))) __bf16 v16bf;
typedef __attribute__((ext_vector_type(8)))  float  v8f;

union BFragU { v16bf v; unsigned u[8]; uint4 q[2]; };
union PackU  { unsigned u; __bf16 h[2]; };

__device__ __forceinline__ float sigmoidf_(float z) { return 1.0f / (1.0f + __expf(-z)); }

template <int O>
__device__ __forceinline__ void cvt8(v16bf& af, float4 a, float4 b) {
    af[O + 0] = (__bf16)a.x; af[O + 1] = (__bf16)a.y;
    af[O + 2] = (__bf16)a.z; af[O + 3] = (__bf16)a.w;
    af[O + 4] = (__bf16)b.x; af[O + 5] = (__bf16)b.y;
    af[O + 6] = (__bf16)b.z; af[O + 7] = (__bf16)b.w;
}

// Build the 16x32 bf16 A-fragment for this lane from a row pointer at K-offset kb.
__device__ __forceinline__ v16bf make_afrag(const float* __restrict__ row, int kb, int half) {
    float4 a0 = *(const float4*)(row + kb + half * 8);
    float4 a1 = *(const float4*)(row + kb + half * 8 + 4);
    float4 a2 = *(const float4*)(row + kb + 16 + half * 8);
    float4 a3 = *(const float4*)(row + kb + 16 + half * 8 + 4);
    v16bf af;
    cvt8<0>(af, a0, a1);
    cvt8<8>(af, a2, a3);
    return af;
}

// ---------------------------------------------------------------------------
// Degree accumulation: deg[d] += 1 for each edge, +1 self loop per node.
// ---------------------------------------------------------------------------
__global__ void k_deg(const long long* __restrict__ dst, float* __restrict__ deg,
                      int E, int n) {
    int t = blockIdx.x * blockDim.x + threadIdx.x;
    if (t < E)            atomicAdd(&deg[(int)dst[t]], 1.0f);
    else if (t < E + n)   atomicAdd(&deg[t - E], 1.0f);
}

__global__ void k_dinv(float* __restrict__ deg, int n) {
    int t = blockIdx.x * blockDim.x + threadIdx.x;
    if (t < n) {
        float d = deg[t];
        deg[t] = (d > 0.0f) ? rsqrtf(d) : 0.0f;
    }
}

// ---------------------------------------------------------------------------
// Pack fp32 weights [K][64] into FRAGMENT-LINEAR bf16 layout:
//   Wf[((ks*4 + c)*32 + lane)*8 + i] = pair( W[K0][n], W[K0+1][n] )
//   K0 = ks*32 + (i>>2)*16 + (lane>>4)*8 + 2*(i&3) ; n = c*16 + (lane&15)
// so each lane's B fragment for (kstep, tile) is 32 contiguous bytes.
// ---------------------------------------------------------------------------
__global__ void k_pack_frag(const float* __restrict__ W, unsigned* __restrict__ Wf,
                            int K) {
    int t = blockIdx.x * blockDim.x + threadIdx.x;
    int total = (K >> 5) * 1024;
    if (t >= total) return;
    int i  = t & 7;
    int L  = (t >> 3) & 31;
    int c  = (t >> 8) & 3;
    int ks = t >> 10;
    int K0 = ks * 32 + (i >> 2) * 16 + (L >> 4) * 8 + 2 * (i & 3);
    int n  = c * 16 + (L & 15);
    PackU p;
    p.h[0] = (__bf16)W[(size_t)K0 * 64 + n];
    p.h[1] = (__bf16)W[(size_t)(K0 + 1) * 64 + n];
    Wf[t] = p.u;
}

// ---------------------------------------------------------------------------
// C[M][64] = A[M][K] (f32 -> bf16 in flight) @ W (bf16, weight-stationary).
// Block = 8 waves; wave = 16x64 stripe (4 accumulator tiles); K stepped by 32.
// All B fragments are hoisted into registers before the K loop, so the steady
// state is A global_load_b128 -> cvt_pk_bf16 -> back-to-back WMMAs.
// ---------------------------------------------------------------------------
template <int NK>
__global__ __launch_bounds__(256) void k_gemm_bf16(
    const float* __restrict__ A, int M,
    const unsigned* __restrict__ Wf, float* __restrict__ C) {
    constexpr int K = NK * 32;
    __shared__ unsigned ldsw[NK * 1024];
    for (int idx = threadIdx.x; idx < NK * 1024; idx += 256)
        ldsw[idx] = Wf[idx];
    __syncthreads();

    int wave = threadIdx.x >> 5;
    int lane = threadIdx.x & 31;
    int half = lane >> 4;
    int m    = lane & 15;
    int trow = blockIdx.x * 128 + wave * 16;
    if (trow >= M) return;  // wave-uniform: EXEC all-ones for WMMA

    // Weight-stationary: all B fragments into registers (NK*4*8 VGPRs).
    BFragU bf[NK][4];
#pragma unroll
    for (int ks = 0; ks < NK; ++ks) {
        const unsigned* base = &ldsw[ks * 1024 + lane * 8];
#pragma unroll
        for (int c = 0; c < 4; ++c) {
            bf[ks][c].q[0] = *(const uint4*)(base + c * 256);
            bf[ks][c].q[1] = *(const uint4*)(base + c * 256 + 4);
        }
    }

    int row  = trow + m;
    int rowc = (row < M) ? row : (M - 1);
    const float* Arow = A + (size_t)rowc * K;

    v8f acc0 = {}, acc1 = {}, acc2 = {}, acc3 = {};
#pragma unroll
    for (int ks = 0; ks < NK; ++ks) {
        v16bf af = make_afrag(Arow, ks * 32, half);
        acc0 = __builtin_amdgcn_wmma_f32_16x16x32_bf16(false, af, false, bf[ks][0].v, (short)0, acc0, false, false);
        acc1 = __builtin_amdgcn_wmma_f32_16x16x32_bf16(false, af, false, bf[ks][1].v, (short)0, acc1, false, false);
        acc2 = __builtin_amdgcn_wmma_f32_16x16x32_bf16(false, af, false, bf[ks][2].v, (short)0, acc2, false, false);
        acc3 = __builtin_amdgcn_wmma_f32_16x16x32_bf16(false, af, false, bf[ks][3].v, (short)0, acc3, false, false);
    }
#pragma unroll
    for (int j = 0; j < 8; ++j) {
        int r = trow + j + 8 * half;  // D layout: VGPR j -> M=j (lanes 0-15) / j+8
        if (r < M) {
            float* crow = C + (size_t)r * 64;
            crow[m]      = acc0[j];
            crow[16 + m] = acc1[j];
            crow[32 + m] = acc2[j];
            crow[48 + m] = acc3[j];
        }
    }
}

// ---------------------------------------------------------------------------
// Normalized scatter-add: agg[d] += Z[s] * dinv[s]*dinv[d]  (+ self loops).
// One 32-lane wave per edge; 2 channels per lane (64 channels).
// ---------------------------------------------------------------------------
__global__ void k_scatter(const long long* __restrict__ src,
                          const long long* __restrict__ dst,
                          const float* __restrict__ dinv,
                          const float* __restrict__ Z,
                          float* __restrict__ agg, int E, int n) {
    int t = blockIdx.x * blockDim.x + threadIdx.x;
    int e = t >> 5, lane = t & 31;
    if (e >= E + n) return;
    int s, d; float nf;
    if (e < E) {
        s = (int)src[e]; d = (int)dst[e];
        nf = dinv[s] * dinv[d];
    } else {
        s = d = e - E;
        float di = dinv[s];
        nf = di * di;
    }
    int ch = lane * 2;
    float2 v = *(const float2*)(Z + (size_t)s * 64 + ch);
    float* o = agg + (size_t)d * 64 + ch;
    atomicAdd(o,     v.x * nf);
    atomicAdd(o + 1, v.y * nf);
}

__global__ void k_bias_relu(const float* __restrict__ in, const float* __restrict__ b,
                            float* __restrict__ out, int total) {
    int t = blockIdx.x * blockDim.x + threadIdx.x;
    if (t < total) out[t] = fmaxf(in[t] + b[t & 63], 0.0f);
}

// ---------------------------------------------------------------------------
// Edge MLP: per wave, 16 edges. A = [h[src] | h[dst]] (16x128) gathered &
// converted to bf16; hid = relu(A@Wm1 + bm1) via 16 WMMAs with all weight
// fragments hoisted to registers; then the 64->1 layer as per-lane dot +
// 16-lane shfl_xor reduce + sigmoid.
// ---------------------------------------------------------------------------
__global__ __launch_bounds__(256) void k_edge_mlp(
    const float* __restrict__ h,
    const long long* __restrict__ src, const long long* __restrict__ dst,
    const unsigned* __restrict__ WfM, const float* __restrict__ bm1,
    const float* __restrict__ Wm2, const float* __restrict__ bm2,
    float* __restrict__ prob, int E) {
    __shared__ unsigned ldsw[4096];  // K=128: 16KB
    for (int idx = threadIdx.x; idx < 4096; idx += 256)
        ldsw[idx] = WfM[idx];
    __syncthreads();

    int wave  = threadIdx.x >> 5;
    int lane  = threadIdx.x & 31;
    int half  = lane >> 4;
    int m     = lane & 15;
    int ebase = (blockIdx.x * 8 + wave) * 16;
    if (ebase >= E) return;  // wave-uniform

    // Weight-stationary: 16 B fragments (128 VGPRs).
    BFragU bf[4][4];
#pragma unroll
    for (int ks = 0; ks < 4; ++ks) {
        const unsigned* base = &ldsw[ks * 1024 + lane * 8];
#pragma unroll
        for (int c = 0; c < 4; ++c) {
            bf[ks][c].q[0] = *(const uint4*)(base + c * 256);
            bf[ks][c].q[1] = *(const uint4*)(base + c * 256 + 4);
        }
    }

    int e  = ebase + m;
    const float* hs = h + (size_t)((int)src[e]) * 64;
    const float* hd = h + (size_t)((int)dst[e]) * 64;

    v8f acc0 = {}, acc1 = {}, acc2 = {}, acc3 = {};
#pragma unroll
    for (int ks = 0; ks < 4; ++ks) {
        const float* hrow = (ks < 2) ? hs : hd;   // K<64 -> src half, else dst
        v16bf af = make_afrag(hrow, (ks & 1) * 32, half);
        acc0 = __builtin_amdgcn_wmma_f32_16x16x32_bf16(false, af, false, bf[ks][0].v, (short)0, acc0, false, false);
        acc1 = __builtin_amdgcn_wmma_f32_16x16x32_bf16(false, af, false, bf[ks][1].v, (short)0, acc1, false, false);
        acc2 = __builtin_amdgcn_wmma_f32_16x16x32_bf16(false, af, false, bf[ks][2].v, (short)0, acc2, false, false);
        acc3 = __builtin_amdgcn_wmma_f32_16x16x32_bf16(false, af, false, bf[ks][3].v, (short)0, acc3, false, false);
    }

    float bb0 = bm1[m],      bb1 = bm1[16 + m], bb2 = bm1[32 + m], bb3 = bm1[48 + m];
    float w0  = Wm2[m],      w1  = Wm2[16 + m], w2  = Wm2[32 + m], w3  = Wm2[48 + m];

    float part[8];
#pragma unroll
    for (int j = 0; j < 8; ++j) {
        float p = fmaxf(acc0[j] + bb0, 0.0f) * w0;
        p      += fmaxf(acc1[j] + bb1, 0.0f) * w1;
        p      += fmaxf(acc2[j] + bb2, 0.0f) * w2;
        p      += fmaxf(acc3[j] + bb3, 0.0f) * w3;
        part[j] = p;
    }
#pragma unroll
    for (int j = 0; j < 8; ++j) {
        float p = part[j];
        p += __shfl_xor(p, 1, 32);
        p += __shfl_xor(p, 2, 32);
        p += __shfl_xor(p, 4, 32);
        p += __shfl_xor(p, 8, 32);
        part[j] = p;
    }
    if (m == 0) {
        float b2v = bm2[0];
#pragma unroll
        for (int j = 0; j < 8; ++j)
            prob[ebase + 8 * half + j] = sigmoidf_(part[j] + b2v);
    }
}

// ---------------------------------------------------------------------------
extern "C" void kernel_launch(void* const* d_in, const int* in_sizes, int n_in,
                              void* d_out, int out_size, void* d_ws, size_t ws_size,
                              hipStream_t stream) {
    const float*     x   = (const float*)d_in[0];
    const long long* ei  = (const long long*)d_in[1];
    const float*     W1  = (const float*)d_in[2];
    const float*     b1  = (const float*)d_in[3];
    const float*     W2  = (const float*)d_in[4];
    const float*     b2  = (const float*)d_in[5];
    const float*     Wm1 = (const float*)d_in[6];
    const float*     bm1 = (const float*)d_in[7];
    const float*     Wm2 = (const float*)d_in[8];
    const float*     bm2 = (const float*)d_in[9];

    const long long* src = ei;            // edge_index[0, :]
    const long long* dst = ei + N_EDGES;  // edge_index[1, :]

    float* out_prob = (float*)d_out;
    float* out_h    = (float*)d_out + N_EDGES;

    char* ws = (char*)d_ws;
    float*    dinv = (float*)(ws);                                   // N*4
    float*    zw   = (float*)(ws + (size_t)524288);                  // N*64*4
    float*    agg  = (float*)(ws + (size_t)524288 + 26214400);       // N*64*4
    unsigned* Wf1  = (unsigned*)(ws + (size_t)524288 + 2 * 26214400);          // 16KB
    unsigned* Wf2  = (unsigned*)(ws + (size_t)524288 + 2 * 26214400 + 16384);  // 8KB
    unsigned* WfM  = (unsigned*)(ws + (size_t)524288 + 2 * 26214400 + 32768);  // 16KB

    // degrees -> dinv
    hipMemsetAsync(dinv, 0, (size_t)N_NODES * 4, stream);
    k_deg<<<(N_EDGES + N_NODES + 255) / 256, 256, 0, stream>>>(dst, dinv, N_EDGES, N_NODES);
    k_dinv<<<(N_NODES + 255) / 256, 256, 0, stream>>>(dinv, N_NODES);

    // pack weights to fragment-linear bf16 layout
    k_pack_frag<<<(4096 + 255) / 256, 256, 0, stream>>>(W1, Wf1, IN_CH);
    k_pack_frag<<<(2048 + 255) / 256, 256, 0, stream>>>(W2, Wf2, HID);
    k_pack_frag<<<(4096 + 255) / 256, 256, 0, stream>>>(Wm1, WfM, 2 * HID);

    int gemm_blocks = (N_NODES + 127) / 128;
    int scat_blocks = (int)(((long long)(N_EDGES + N_NODES) * 32 + 255) / 256);
    int br_blocks   = (N_NODES * 64 + 255) / 256;

    // GCN layer 1 (K=128)
    k_gemm_bf16<4><<<gemm_blocks, 256, 0, stream>>>(x, N_NODES, Wf1, zw);
    hipMemsetAsync(agg, 0, (size_t)N_NODES * 64 * 4, stream);
    k_scatter<<<scat_blocks, 256, 0, stream>>>(src, dst, dinv, zw, agg, N_EDGES, N_NODES);
    k_bias_relu<<<br_blocks, 256, 0, stream>>>(agg, b1, agg, N_NODES * 64);

    // GCN layer 2 (K=64)
    k_gemm_bf16<2><<<gemm_blocks, 256, 0, stream>>>(agg, N_NODES, Wf2, zw);
    hipMemsetAsync(agg, 0, (size_t)N_NODES * 64 * 4, stream);
    k_scatter<<<scat_blocks, 256, 0, stream>>>(src, dst, dinv, zw, agg, N_EDGES, N_NODES);
    k_bias_relu<<<br_blocks, 256, 0, stream>>>(agg, b2, out_h, N_NODES * 64);

    // edge MLP -> probabilities
    k_edge_mlp<<<(N_EDGES / 16 + 7) / 8, 256, 0, stream>>>(
        out_h, src, dst, WfM, bm1, Wm2, bm2, out_prob, N_EDGES);
}
